// DeformConv2d_15719580304132
// MI455X (gfx1250) — matrix operand order repaired and checked by reading, not compiled, not run on hardware
//
#include <hip/hip_runtime.h>

typedef __attribute__((ext_vector_type(8)))  _Float16 v8h;
typedef __attribute__((ext_vector_type(16))) _Float16 v16h;
typedef __attribute__((ext_vector_type(8)))  float    v8f;

#define N_IMG  8
#define CIN    64
#define COUT   64
#define HH     128
#define WW     128
#define KK     9
#define KDIM   576   // CIN * KK
#define KSTEPS 18    // KDIM / 32
#define ASTRIDE 584  // padded LDS row stride in halves (1168 B, 16B aligned, conflict-free)

// ---------------------------------------------------------------------------
// Pre-kernel: repack fp32 weights [Cout][Cin*9] into f16 WMMA A-fragment order.
// Flat layout: wh[(((cb*18 + ks)*32) + lane)*16 + j]
//   lane<16 : M=lane,    K(j) = j<8 ? j   : j+8    (K in {0..7, 16..23})
//   lane>=16: M=lane-16, K(j) = j<8 ? j+8 : j+16   (K in {8..15, 24..31})
// ---------------------------------------------------------------------------
__global__ void repack_weights_k(const float* __restrict__ w,
                                 _Float16* __restrict__ wh) {
  int i = blockIdx.x * 256 + threadIdx.x;
  if (i >= 4 * KSTEPS * 32 * 16) return;
  int j    = i & 15;
  int lane = (i >> 4) & 31;
  int blk  = i >> 9;            // cb*18 + ks
  int ks   = blk % KSTEPS;
  int cb   = blk / KSTEPS;
  int m    = lane & 15;
  int K    = (lane < 16) ? ((j < 8) ? j : j + 8)
                         : ((j < 8) ? j + 8 : j + 16);
  int kdim = ks * 32 + K;       // < 576 always (576 = 18*32)
  int cout = cb * 16 + m;
  wh[i] = (_Float16)w[cout * KDIM + kdim];
}

// ---------------------------------------------------------------------------
// Main kernel: one block = 16 output pixels (same n, same y), 128 threads.
//   Phase 1: bilinear gather + modulation -> f16 im2col tile in LDS (16 x 576)
//   Phase 2: 4 waves, wave cb computes couts [16cb,16cb+16) via 18 WMMAs
// ---------------------------------------------------------------------------
__global__ __launch_bounds__(128) void dcnv2_wmma_k(
    const float* __restrict__ x, const float* __restrict__ off,
    const float* __restrict__ mask, const _Float16* __restrict__ wh,
    const float* __restrict__ bias, float* __restrict__ out) {
  __shared__ __align__(16) _Float16 Alds[16][ASTRIDE];

  const int tid = threadIdx.x;
  const int bid = blockIdx.x;
  const int x0  = (bid & 7) * 16;        // tile base x
  const int y   = (bid >> 3) & 127;      // row
  const int n   = bid >> 10;             // image

  // ---------------- Phase 1: deformable sampling ----------------
  const int p  = tid & 15;               // pixel within tile
  const int g  = tid >> 4;               // channel group 0..7 (8 channels each)
  const int xw = x0 + p;

  for (int kk = 0; kk < KK; ++kk) {
    const int ky = kk / 3, kx = kk % 3;
    const float dy = off[(((size_t)n * 18 + 2 * kk) * HH + y) * WW + xw];
    const float dx = off[(((size_t)n * 18 + 2 * kk + 1) * HH + y) * WW + xw];
    const float mv = mask[(((size_t)n * 9 + kk) * HH + y) * WW + xw];

    const float py = (float)(y - 1 + ky) + dy;
    const float px = (float)(xw - 1 + kx) + dx;
    const bool valid = (py > -1.f) && (py < (float)HH) &&
                       (px > -1.f) && (px < (float)WW);

    const float fy0 = floorf(py), fx0 = floorf(px);
    const float wy = py - fy0, wx = px - fx0;
    const int y0 = (int)fy0, xx0 = (int)fx0;
    const int y1 = y0 + 1,  xx1 = xx0 + 1;

    const bool oky0 = valid && (y0 >= 0) && (y0 < HH);
    const bool oky1 = valid && (y1 >= 0) && (y1 < HH);
    const bool okx0 = valid && (xx0 >= 0) && (xx0 < WW);
    const bool okx1 = valid && (xx1 >= 0) && (xx1 < WW);

    const float w00 = (1.f - wy) * (1.f - wx);
    const float w01 = (1.f - wy) * wx;
    const float w10 = wy * (1.f - wx);
    const float w11 = wy * wx;

    const int y0c = min(max(y0, 0), HH - 1), y1c = min(max(y1, 0), HH - 1);
    const int x0c = min(max(xx0, 0), WW - 1), x1c = min(max(xx1, 0), WW - 1);

    #pragma unroll
    for (int ci = 0; ci < 8; ++ci) {
      const int c = g * 8 + ci;
      const float* img = x + ((size_t)(n * CIN + c)) * (HH * WW);
      float v00 = (oky0 && okx0) ? img[y0c * WW + x0c] : 0.f;
      float v01 = (oky0 && okx1) ? img[y0c * WW + x1c] : 0.f;
      float v10 = (oky1 && okx0) ? img[y1c * WW + x0c] : 0.f;
      float v11 = (oky1 && okx1) ? img[y1c * WW + x1c] : 0.f;
      float val = (w00 * v00 + w01 * v01 + w10 * v10 + w11 * v11) * mv;
      Alds[p][c * 9 + kk] = (_Float16)val;   // K-dim index = c*9 + kk
    }
  }
  __syncthreads();

  // ---------------- Phase 2: WMMA GEMM ----------------
  const int lane = tid & 31;
  const int cb   = tid >> 5;          // wave id -> cout block
  const int m16  = lane & 15;
  const bool hi  = lane >= 16;

  v8f acc = {};
  const _Float16* wbase = wh + ((size_t)cb * KSTEPS * 32 + lane) * 16;

  #pragma unroll
  for (int ks = 0; ks < KSTEPS; ++ks) {
    // A = weight fragment, pre-packed: one coalesced 32B load per lane
    v16h af = *(const v16h*)(wbase + (size_t)ks * 32 * 16);
    // B = im2col fragment: lane holds pixel m16, K = k0..k0+7 and k0+16..k0+23
    const int k0 = ks * 32 + (hi ? 8 : 0);
    v8h b0 = *(const v8h*)&Alds[m16][k0];
    v8h b1 = *(const v8h*)&Alds[m16][k0 + 16];
    v16h bf = __builtin_shufflevector(b0, b1, 0, 1, 2, 3, 4, 5, 6, 7,
                                      8, 9, 10, 11, 12, 13, 14, 15);
    acc = __builtin_amdgcn_wmma_f32_16x16x32_f16(
        /*neg_a=*/false, af, /*neg_b=*/false, bf,
        /*c_mod=*/(short)0, acc, /*reuse_a=*/false, /*reuse_b=*/false);
  }

  // ---------------- bias + coalesced store ----------------
  // D layout: lanes 0-15: (M=r, N=lane); lanes 16-31: (M=8+r, N=lane-16)
  #pragma unroll
  for (int r = 0; r < 8; ++r) {
    const int cout = cb * 16 + r + (hi ? 8 : 0);
    const float v = acc[r] + bias[cout];
    out[(((size_t)n * COUT + cout) * HH + y) * WW + x0 + m16] = v;
  }
}

// ---------------------------------------------------------------------------
extern "C" void kernel_launch(void* const* d_in, const int* in_sizes, int n_in,
                              void* d_out, int out_size, void* d_ws,
                              size_t ws_size, hipStream_t stream) {
  const float* x      = (const float*)d_in[0];
  const float* off    = (const float*)d_in[1];
  const float* mask   = (const float*)d_in[2];
  const float* weight = (const float*)d_in[3];
  const float* bias   = (const float*)d_in[4];
  float* out          = (float*)d_out;
  _Float16* wh        = (_Float16*)d_ws;   // 4*18*32*16 halves = 73728 B

  repack_weights_k<<<144, 256, 0, stream>>>(weight, wh);

  const int tiles = N_IMG * HH * (WW / 16);  // 8192
  dcnv2_wmma_k<<<tiles, 128, 0, stream>>>(x, off, mask, wh, bias, out);
}
